// RNFCell_48043504173109
// MI455X (gfx1250) — matrix operand order repaired and checked by reading, not compiled, not run on hardware
//
#include <hip/hip_runtime.h>
#include <hip/hip_bf16.h>

typedef __attribute__((ext_vector_type(16))) _Float16 v16h;
typedef __attribute__((ext_vector_type(8)))  _Float16 v8h;
typedef __attribute__((ext_vector_type(4)))  _Float16 v4h;
typedef __attribute__((ext_vector_type(8)))  float    v8f;
typedef __attribute__((ext_vector_type(4)))  unsigned int v4u;
typedef __attribute__((ext_vector_type(8)))  int      v8i;
typedef __attribute__((ext_vector_type(4)))  int      v4i;

#define BATCH 32768
#define HDIM  256
#define NIN   64
#define NOUT  8
#define DROW  72          // data row = NIN + NOUT

// Each wave owns M=32 rows (two 16-row WMMA tiles) => every B fragment load
// feeds 2 WMMAs, halving the dominant L2 operand traffic.
#define WAVES     4
#define ROWS_W    32
#define BLOCK_M   (WAVES * ROWS_W)   // 128

// LDS strides (elements)
#define HSTR  264         // 256 + 8 halfword pad (conflict-free b128 A-loads)
#define CSTR  260         // 256 + 4 float pad   (matches TDM pad 256dw + 4dw)
#define XSTR  72
#define OSTR  32

// per-wave LDS byte offsets (all 16B aligned)
#define OFF_HA  0                     // 32 x HSTR f16  = 16896
#define OFF_HB  16896                 // 32 x HSTR f16  = 16896
#define OFF_CB  33792                 // 32 x CSTR f32  = 33280
#define OFF_XB  67072                 // 32 x XSTR f16  = 4608
#define OFF_OB  71680                 // 32 x OSTR f16  = 2048
#define OFF_FL  73728                 // 32 x 2 f32     = 256
#define WAVE_LDS 73984                // x4 waves = 295,936 B  (< 320 KB WGP)

// workspace f16 fragment offsets (halfword units)
#define WOFF_WR1 0
#define WOFF_WR2 262144
#define WOFF_WR3 524288
#define WOFF_WK2 786432
#define WOFF_WK3 851968
#define WOFF_WD  884736
#define WOFF_WMS 950272

__device__ __forceinline__ float sigm_f(float x)  { return 1.0f / (1.0f + __expf(-x)); }
__device__ __forceinline__ float elu_f(float x)   { return x > 0.0f ? x : (__expf(x) - 1.0f); }
__device__ __forceinline__ float softp_f(float x) { return logf(1.0f + __expf(x)); }

// ---------------------------------------------------------------------------
// Tensor Data Mover: 2D f32 tile (td0_dw dwords x nrows) global -> LDS,
// optional LDS row padding (pad codes per CDNA5 D# spec).
// clang-23 toolchain: 6-arg builtin (g0 v4u, g1 v8i, g2 v4i, g3 v4i, vx v8i, cpol).
// ---------------------------------------------------------------------------
__device__ __forceinline__ void tdm_load_2d_f32(unsigned int lds_byte, const void* gsrc,
                                                unsigned int td0_dw, unsigned int nrows,
                                                unsigned int row_stride_dw,
                                                unsigned int pad_interval_code,
                                                unsigned int pad_amount_code,
                                                bool pad_en) {
  unsigned long long ga = (unsigned long long)(uintptr_t)gsrc;
  v4u g0;
  g0[0] = 1u;                                         // count=1, user mode
  g0[1] = lds_byte;                                   // lds_addr
  g0[2] = (unsigned int)(ga & 0xFFFFFFFFull);         // global_addr[31:0]
  g0[3] = (unsigned int)((ga >> 32) & 0x01FFFFFFull)  // global_addr[56:32]
        | (2u << 30);                                 // type = 2 ("image")
  unsigned int w0 = (2u << 16);                       // data_size = 4 bytes
  if (pad_en) w0 |= (1u << 20) | (pad_interval_code << 22) | (pad_amount_code << 25);
  v8i g1;
  g1[0] = (int)w0;
  g1[1] = (int)((td0_dw & 0xFFFFu) << 16);            // tensor_dim0[15:0]
  g1[2] = (int)(((nrows & 0xFFFFu) << 16) | ((td0_dw >> 16) & 0xFFFFu)); // tdim1 | tdim0 hi
  g1[3] = (int)((td0_dw & 0xFFFFu) << 16);            // tile_dim0 (| tdim1 hi = 0)
  g1[4] = (int)(nrows & 0xFFFFu);                     // tile_dim1 (tile_dim2 = 0)
  g1[5] = (int)row_stride_dw;                         // tensor_dim0_stride[31:0]
  g1[6] = 0;
  g1[7] = 0;
  v4i z4 = {0, 0, 0, 0};                              // 2D: groups 2/3 unused
  v8i z8 = {0, 0, 0, 0, 0, 0, 0, 0};
  __builtin_amdgcn_tensor_load_to_lds(g0, g1, z4, z4, z8, 0);
}

// ---------------------------------------------------------------------------
// Weight packing: KxN f32 -> f16 WMMA-B fragments (lane-swizzled, contiguous).
// frag f = nt*nkt + kt; lane l -> col n = nt*16+(l&15); halfword e ->
// k = kt*32 + (e<8 ? 8*(l>>4) + e : 16 + 8*(l>>4) + (e-8)).
// ---------------------------------------------------------------------------
__global__ void pack_b(const float* __restrict__ W, _Float16* __restrict__ out,
                       int K, int N, int nkt, int total) {
  int idx = blockIdx.x * blockDim.x + threadIdx.x;
  if (idx >= total) return;
  int e    = idx & 15;
  int lane = (idx >> 4) & 31;
  int f    = idx >> 9;
  int nt   = f / nkt;
  int kt   = f - nt * nkt;
  int n = nt * 16 + (lane & 15);
  int k = kt * 32 + ((e < 8) ? 8 * (lane >> 4) + e : 16 + 8 * (lane >> 4) + (e - 8));
  float v = (k < K && n < N) ? W[(size_t)k * N + n] : 0.0f;
  out[idx] = (_Float16)v;
}

// Fused [Wm | Ws] head weight: K=256, N=16 (cols 0..7 = Wm, 8..15 = Ws), nkt=8.
__global__ void pack_b_ms(const float* __restrict__ Wm, const float* __restrict__ Ws,
                          _Float16* __restrict__ out) {
  int idx = blockIdx.x * blockDim.x + threadIdx.x;
  if (idx >= 4096) return;
  int e    = idx & 15;
  int lane = (idx >> 4) & 31;
  int kt   = idx >> 9;
  int n = lane & 15;
  int k = kt * 32 + ((e < 8) ? 8 * (lane >> 4) + e : 16 + 8 * (lane >> 4) + (e - 8));
  float v = (n < 8) ? Wm[(size_t)k * NOUT + n] : Ws[(size_t)k * NOUT + (n - 8)];
  out[idx] = (_Float16)v;
}

__device__ __forceinline__ v16h load_a_frag(const _Float16* arow, int kbase) {
  v8h lo = *(const v8h*)(arow + kbase);
  v8h hi = *(const v8h*)(arow + kbase + 16);
  v16h r;
#pragma unroll
  for (int i = 0; i < 8; ++i) { r[i] = lo[i]; r[i + 8] = hi[i]; }
  return r;
}

// acc[g][mt] += A(mt, LDS) x Bfrag(global, L2-resident) for the 4 gate tiles
// of hidden chunk jt; each B fragment is reused for both 16-row A tiles.
__device__ __forceinline__ void gate_gemm2(const _Float16* abuf, int astride, int nkt,
                                           const _Float16* wfrag, int jt, int lane,
                                           v8f (*acc)[2]) {
  const _Float16* arow0 = abuf + (lane & 15) * astride + 8 * (lane >> 4);
  const _Float16* arow1 = arow0 + 16 * astride;
  for (int kt = 0; kt < nkt; ++kt) {
    v16h a0 = load_a_frag(arow0, kt * 32);
    v16h a1 = load_a_frag(arow1, kt * 32);
#pragma unroll
    for (int g = 0; g < 4; ++g) {
      const _Float16* bp = wfrag + (((size_t)(g * 16 + jt) * nkt + kt) * 32 + lane) * 16;
      v16h b = *(const v16h*)bp;
      acc[g][0] = __builtin_amdgcn_wmma_f32_16x16x32_f16(false, a0, false, b,
                                                         (short)0, acc[g][0], false, false);
      acc[g][1] = __builtin_amdgcn_wmma_f32_16x16x32_f16(false, a1, false, b,
                                                         (short)0, acc[g][1], false, false);
    }
  }
}

// Single N-tile GEMM for both M tiles (decoder / heads); B reused 2x.
__device__ __forceinline__ void gemm_1nt2(const _Float16* abuf, int astride, int nkt,
                                          const _Float16* wfrag, int nt, int lane,
                                          v8f* acc) {
  const _Float16* arow0 = abuf + (lane & 15) * astride + 8 * (lane >> 4);
  const _Float16* arow1 = arow0 + 16 * astride;
  for (int kt = 0; kt < nkt; ++kt) {
    v16h a0 = load_a_frag(arow0, kt * 32);
    v16h a1 = load_a_frag(arow1, kt * 32);
    const _Float16* bp = wfrag + (((size_t)nt * nkt + kt) * 32 + lane) * 16;
    v16h b = *(const v16h*)bp;
    acc[0] = __builtin_amdgcn_wmma_f32_16x16x32_f16(false, a0, false, b,
                                                    (short)0, acc[0], false, false);
    acc[1] = __builtin_amdgcn_wmma_f32_16x16x32_f16(false, a1, false, b,
                                                    (short)0, acc[1], false, false);
  }
}

__global__ __launch_bounds__(128)
void lstm_fused(const float* __restrict__ data, const float* __restrict__ flags,
                const float* __restrict__ h0,   const float* __restrict__ c0,
                const float* __restrict__ b1,   const float* __restrict__ b2,
                const float* __restrict__ b3,   const float* __restrict__ bd,
                const float* __restrict__ bm,   const float* __restrict__ bs,
                const _Float16* __restrict__ ws,
                float* __restrict__ out_ms, float* __restrict__ out_h,
                float* __restrict__ out_c) {
  extern __shared__ __align__(32) char smem[];
  const int lane = threadIdx.x & 31;
  const int wid  = threadIdx.x >> 5;
  const int rowbase = blockIdx.x * BLOCK_M + wid * ROWS_W;

  char* wbase = smem + wid * WAVE_LDS;
  _Float16* hA = (_Float16*)(wbase + OFF_HA);
  _Float16* hB = (_Float16*)(wbase + OFF_HB);
  float*    cb = (float*)   (wbase + OFF_CB);
  _Float16* xb = (_Float16*)(wbase + OFF_XB);
  _Float16* ob = (_Float16*)(wbase + OFF_OB);
  float*    fl = (float*)   (wbase + OFF_FL);

  const _Float16* wr1 = ws + WOFF_WR1;
  const _Float16* wr2 = ws + WOFF_WR2;
  const _Float16* wr3 = ws + WOFF_WR3;
  const _Float16* wk2 = ws + WOFF_WK2;
  const _Float16* wk3 = ws + WOFF_WK3;
  const _Float16* wd  = ws + WOFF_WD;
  const _Float16* wms = ws + WOFF_WMS;

  // ---- c0 -> cb via Tensor Data Mover (32 rows x 1KB, LDS rows padded
  //      256dw + 4dw  ==  CSTR):  pad_interval code 7 (=256 dw), amount 3 (=4 dw)
  tdm_load_2d_f32((unsigned int)(uintptr_t)cb, c0 + (size_t)rowbase * HDIM,
                  /*td0_dw=*/HDIM, /*nrows=*/ROWS_W, /*row_stride_dw=*/HDIM,
                  /*pad_interval=*/7u, /*pad_amount=*/3u, /*pad_en=*/true);

  // ---- stage-in: h0 -> hA (f16), x -> xb, obs -> ob, flags -> fl
  fl[lane]      = flags[(size_t)(rowbase + (lane >> 1)) * 2 + (lane & 1)];
  fl[lane + 32] = flags[(size_t)(rowbase + 16 + (lane >> 1)) * 2 + (lane & 1)];

  for (int t = lane * 4; t < ROWS_W * 256; t += 128) {
    int r = t >> 8, c = t & 255;
    float4 hv = *(const float4*)(h0 + (size_t)(rowbase + r) * HDIM + c);
    v4h hh = { (_Float16)hv.x, (_Float16)hv.y, (_Float16)hv.z, (_Float16)hv.w };
    *(v4h*)(hA + r * HSTR + c) = hh;
  }
  for (int t = lane * 4; t < ROWS_W * 64; t += 128) {
    int r = t >> 6, c = t & 63;
    float4 xv = *(const float4*)(data + (size_t)(rowbase + r) * DROW + c);
    v4h xh = { (_Float16)xv.x, (_Float16)xv.y, (_Float16)xv.z, (_Float16)xv.w };
    *(v4h*)(xb + r * XSTR + c) = xh;
  }
  for (int t = lane; t < ROWS_W * 32; t += 32) {
    int r = t >> 5, c = t & 31;
    float v = (c < 8) ? data[(size_t)(rowbase + r) * DROW + NIN + c] : 0.0f;
    ob[r * OSTR + c] = (_Float16)v;
  }
  __builtin_amdgcn_s_wait_tensorcnt(0);   // cb tile landed
  __syncthreads();

  const int col16 = lane & 15;
  const int kh    = lane >> 4;

  // ================= stage 1: z = h0 @ Wr1 + b1 (x is zeros) =================
  for (int jt = 0; jt < 16; ++jt) {
    v8f acc[4][2] = {};
    gate_gemm2(hA, HSTR, 8, wr1, jt, lane, acc);
    int hcol = jt * 16 + col16;
    float bi = b1[hcol], bfv = b1[HDIM + hcol], bg = b1[2 * HDIM + hcol], bo = b1[3 * HDIM + hcol];
#pragma unroll
    for (int mt = 0; mt < 2; ++mt) {
#pragma unroll
      for (int v = 0; v < 8; ++v) {
        int M = mt * 16 + v + 8 * kh;
        float cold = cb[M * CSTR + hcol];
        float cn = sigm_f(acc[1][mt][v] + bfv) * cold + sigm_f(acc[0][mt][v] + bi) * elu_f(acc[2][mt][v] + bg);
        float hn = sigm_f(acc[3][mt][v] + bo) * elu_f(cn);
        cb[M * CSTR + hcol] = cn;               // c1
        hB[M * HSTR + hcol] = (_Float16)hn;     // h1
      }
    }
  }
  __syncthreads();

  // ============ stage 2: z = x @ Wk2 + h1 @ Wr2 + b2; blend keep2 ============
  for (int jt = 0; jt < 16; ++jt) {
    v8f acc[4][2] = {};
    gate_gemm2(hB, HSTR, 8, wr2, jt, lane, acc);
    gate_gemm2(xb, XSTR, 2, wk2, jt, lane, acc);
    int hcol = jt * 16 + col16;
    float bi = b2[hcol], bfv = b2[HDIM + hcol], bg = b2[2 * HDIM + hcol], bo = b2[3 * HDIM + hcol];
#pragma unroll
    for (int mt = 0; mt < 2; ++mt) {
#pragma unroll
      for (int v = 0; v < 8; ++v) {
        int M = mt * 16 + v + 8 * kh;
        float k2   = fl[2 * M];
        float cold = cb[M * CSTR + hcol];               // c1
        float hprv = (float)hB[M * HSTR + hcol];        // h1
        float cn = sigm_f(acc[1][mt][v] + bfv) * cold + sigm_f(acc[0][mt][v] + bi) * elu_f(acc[2][mt][v] + bg);
        float hn = sigm_f(acc[3][mt][v] + bo) * elu_f(cn);
        cb[M * CSTR + hcol] = cold + k2 * (cn - cold);  // c2s
        hA[M * HSTR + hcol] = (_Float16)(hprv + k2 * (hn - hprv)); // h2s (== lstm_op)
      }
    }
  }
  __syncthreads();

  // ===== stage 3: z = obs @ Wk3 + h2s @ Wr3 + b3; blend keep3 -> outputs =====
  for (int jt = 0; jt < 16; ++jt) {
    v8f acc[4][2] = {};
    gate_gemm2(hA, HSTR, 8, wr3, jt, lane, acc);
    gate_gemm2(ob, OSTR, 1, wk3, jt, lane, acc);
    int hcol = jt * 16 + col16;
    float bi = b3[hcol], bfv = b3[HDIM + hcol], bg = b3[2 * HDIM + hcol], bo = b3[3 * HDIM + hcol];
#pragma unroll
    for (int mt = 0; mt < 2; ++mt) {
#pragma unroll
      for (int v = 0; v < 8; ++v) {
        int M = mt * 16 + v + 8 * kh;
        float k3   = fl[2 * M + 1];
        float cold = cb[M * CSTR + hcol];               // c2s
        float hprv = (float)hA[M * HSTR + hcol];        // h2s
        float cn = sigm_f(acc[1][mt][v] + bfv) * cold + sigm_f(acc[0][mt][v] + bi) * elu_f(acc[2][mt][v] + bg);
        float hn = sigm_f(acc[3][mt][v] + bo) * elu_f(cn);
        size_t go = (size_t)(rowbase + M) * HDIM + hcol;
        out_c[go] = cold + k3 * (cn - cold);            // c3s
        out_h[go] = hprv + k3 * (hn - hprv);            // h3s
      }
    }
  }
  __syncthreads();

  // ================= decoder: hid = elu(h2s @ Wd + bd) -> hB =================
  for (int jt = 0; jt < 16; ++jt) {
    v8f acc[2] = {};
    gemm_1nt2(hA, HSTR, 8, wd, jt, lane, acc);
    int hcol = jt * 16 + col16;
    float bb = bd[hcol];
#pragma unroll
    for (int mt = 0; mt < 2; ++mt)
#pragma unroll
      for (int v = 0; v < 8; ++v) {
        int M = mt * 16 + v + 8 * kh;
        hB[M * HSTR + hcol] = (_Float16)elu_f(acc[mt][v] + bb);
      }
  }
  __syncthreads();

  // ======== heads: out[:,0:8]=hid@Wm+bm ; out[:,8:16]=softplus(hid@Ws+bs) =====
  {
    v8f acc[2] = {};
    gemm_1nt2(hB, HSTR, 8, wms, 0, lane, acc);
    float bb = (col16 < 8) ? bm[col16] : bs[col16 - 8];
#pragma unroll
    for (int mt = 0; mt < 2; ++mt)
#pragma unroll
      for (int v = 0; v < 8; ++v) {
        int M = mt * 16 + v + 8 * kh;
        float z = acc[mt][v] + bb;
        float r = (col16 < 8) ? z : softp_f(z);
        out_ms[(size_t)(rowbase + M) * 16 + col16] = r;
      }
  }
}

extern "C" void kernel_launch(void* const* d_in, const int* in_sizes, int n_in,
                              void* d_out, int out_size, void* d_ws, size_t ws_size,
                              hipStream_t stream) {
  const float* data  = (const float*)d_in[0];
  const float* flags = (const float*)d_in[1];
  const float* h0    = (const float*)d_in[2];
  const float* c0    = (const float*)d_in[3];
  // d_in[4] = Wk1 (multiplied by zeros -> unused)
  const float* Wr1   = (const float*)d_in[5];
  const float* b1    = (const float*)d_in[6];
  const float* Wk2   = (const float*)d_in[7];
  const float* Wr2   = (const float*)d_in[8];
  const float* b2    = (const float*)d_in[9];
  const float* Wk3   = (const float*)d_in[10];
  const float* Wr3   = (const float*)d_in[11];
  const float* b3    = (const float*)d_in[12];
  const float* Wd    = (const float*)d_in[13];
  const float* bd    = (const float*)d_in[14];
  const float* Wm    = (const float*)d_in[15];
  const float* bm    = (const float*)d_in[16];
  const float* Ws    = (const float*)d_in[17];
  const float* bs    = (const float*)d_in[18];

  _Float16* ws = (_Float16*)d_ws;
  float* out_ms = (float*)d_out;                 // (B,16)
  float* out_h  = out_ms + (size_t)BATCH * 16;   // (B,256)
  float* out_c  = out_h + (size_t)BATCH * HDIM;  // (B,256)

  // -- weight packing (f32 -> f16 WMMA-B fragments), all on stream --
  auto launch_pack = [&](const float* W, int off, int K, int N, int nkt) {
    int total = (N / 16) * nkt * 512;
    pack_b<<<(total + 255) / 256, 256, 0, stream>>>(W, ws + off, K, N, nkt, total);
  };
  launch_pack(Wr1, WOFF_WR1, 256, 1024, 8);
  launch_pack(Wr2, WOFF_WR2, 256, 1024, 8);
  launch_pack(Wr3, WOFF_WR3, 256, 1024, 8);
  launch_pack(Wk2, WOFF_WK2, 64, 1024, 2);
  launch_pack(Wk3, WOFF_WK3, 8, 1024, 1);   // K padded 8 -> 32 with zeros
  launch_pack(Wd,  WOFF_WD,  256, 256, 8);
  pack_b_ms<<<4096 / 256, 256, 0, stream>>>(Wm, Ws, ws + WOFF_WMS);

  // -- fused LSTM pipeline: 4 waves/block, 32 rows/wave, 256 blocks --
  dim3 grid(BATCH / BLOCK_M), block(WAVES * 32);
  size_t smem = (size_t)WAVES * WAVE_LDS;   // 295,936 B dynamic LDS (< 320 KB/WGP)
  lstm_fused<<<grid, block, smem, stream>>>(data, flags, h0, c0,
                                            b1, b2, b3, bd, bm, bs,
                                            ws, out_ms, out_h, out_c);
}
